// AdvancedNeuroplasticityLayer_21912923144260
// MI455X (gfx1250) — matrix operand order repaired and checked by reading, not compiled, not run on hardware
//
#include <hip/hip_runtime.h>
#include <hip/hip_bf16.h>
#include <math.h>

// Problem constants (match reference)
#define BB 64
#define II 2048
#define OO 2048
#define SS 10
#define CC 512

typedef __attribute__((ext_vector_type(16))) __bf16 v16bf;
typedef __attribute__((ext_vector_type(8)))  __bf16 v8bf;
typedef __attribute__((ext_vector_type(8)))  float  v8f;
typedef __attribute__((ext_vector_type(4)))  float  v4f;   // native vector for NT loads

// ---------------------------------------------------------------------------
// A-tile loader from pre-packed bf16 x (row-major [B, I] bf16).
// bf16 A k order used here, same as B tile (16x32, M x K), lane m = lane%16, half = lane/16:
//   element j<8  -> k = kb + half*16 + j          (contiguous run of 8)
//   element j>=8 -> k = kb + half*16 + 8 + (j-8)  (contiguous run of 8)
// Two 16-byte loads, zero conversion work.
// ---------------------------------------------------------------------------
__device__ __forceinline__ v16bf load_a_bf(const __bf16* __restrict__ xrow,
                                           int kb, int half) {
  union {
    v16bf v;
    struct { v8bf lo, hi; } p;
  } u;
  u.p.lo = *(const v8bf*)(xrow + kb + half * 16);
  u.p.hi = *(const v8bf*)(xrow + kb + half * 16 + 8);
  return u.v;
}

__device__ __forceinline__ v8f wmma_bf16(v16bf a, v16bf b, v8f c) {
  return __builtin_amdgcn_wmma_f32_16x16x32_bf16(false, a, false, b,
                                                 (short)0, c, false, false);
}

// ---------------------------------------------------------------------------
// Kernel 0: pack x [B, I] f32 -> bf16 row-major (done once per launch).
// ---------------------------------------------------------------------------
__global__ void pack_x_kernel(const float* __restrict__ x,
                              __bf16* __restrict__ xb) {
  int i = (blockIdx.x * blockDim.x + threadIdx.x) * 8;  // B*I = 131072, /8 = 16384 threads
  v4f f0 = *(const v4f*)(x + i);
  v4f f1 = *(const v4f*)(x + i + 4);
  v8bf t;
  t[0] = (__bf16)f0.x; t[1] = (__bf16)f0.y; t[2] = (__bf16)f0.z; t[3] = (__bf16)f0.w;
  t[4] = (__bf16)f1.x; t[5] = (__bf16)f1.y; t[6] = (__bf16)f1.z; t[7] = (__bf16)f1.w;
  *(v8bf*)(xb + i) = t;
}

// ---------------------------------------------------------------------------
// Kernel 1: ctx_mean[b] = mean(context[b, :])
// ---------------------------------------------------------------------------
__global__ void ctx_mean_kernel(const float* __restrict__ context,
                                float* __restrict__ ctx_mean) {
  int b = threadIdx.x;
  if (b < BB) {
    float s = 0.f;
    const float* row = context + (size_t)b * CC;
    #pragma unroll 8
    for (int c = 0; c < CC; ++c) s += row[c];
    ctx_mean[b] = s * (1.0f / CC);
  }
}

// ---------------------------------------------------------------------------
// Kernel 2: mean_astro[o] = mean_b( thresholded sigmoid(ctx_mean[b]*act[o]) )
// ---------------------------------------------------------------------------
__global__ void astro_kernel(const float* __restrict__ ctx_mean,
                             const float* __restrict__ act,
                             const float* __restrict__ thr,
                             float* __restrict__ mean_astro) {
  int o = blockIdx.x * blockDim.x + threadIdx.x;
  if (o < OO) {
    float ao = act[o], to = thr[o];
    float s = 0.f;
    #pragma unroll 8
    for (int b = 0; b < BB; ++b) {
      float z = ctx_mean[b] * ao;
      float a = 1.0f / (1.0f + expf(-z));
      s += (a > to) ? a : 0.f;
    }
    mean_astro[o] = s * (1.0f / BB);
  }
}

// ---------------------------------------------------------------------------
// Kernel 3: synaptic GEMM
//   syn[b,o] = mean_astro[o] * sum_k x[b,k]*eff_w[o,k] + bias[o]
//   eff_w = weight * (|weight|*conn_strength > prune_thr)
//   32 blocks x 128 threads; wave -> one 16-col N-tile, all 4 M-tiles.
// ---------------------------------------------------------------------------
__global__ void syn_gemm_kernel(const __bf16* __restrict__ xb,
                                const float* __restrict__ weight,
                                const float* __restrict__ cstr,
                                const float* __restrict__ prune_thr,
                                const float* __restrict__ bias,
                                const float* __restrict__ mean_astro,
                                float* __restrict__ syn) {
  const int lane = threadIdx.x & 31;
  const int wv   = threadIdx.x >> 5;
  const int col  = lane & 15;
  const int half = lane >> 4;
  const int o    = (blockIdx.x * 4 + wv) * 16 + col;
  const float th = prune_thr[0];

  const float* wrow = weight + (size_t)o * II;
  const float* crow = cstr   + (size_t)o * II;

  v8f acc0 = {}, acc1 = {}, acc2 = {}, acc3 = {};

  for (int kb = 0; kb < II; kb += 32) {
    // B tile: lane holds column o, k = kb + half*16 + j (16 contiguous f32)
    v16bf bt;
    const int kk = kb + half * 16;
    #pragma unroll
    for (int q = 0; q < 4; ++q) {
      v4f wf = __builtin_nontemporal_load((const v4f*)(wrow + kk + q * 4));
      v4f cf = __builtin_nontemporal_load((const v4f*)(crow + kk + q * 4));
      float w0 = (fabsf(wf.x) * cf.x > th) ? wf.x : 0.f;
      float w1 = (fabsf(wf.y) * cf.y > th) ? wf.y : 0.f;
      float w2 = (fabsf(wf.z) * cf.z > th) ? wf.z : 0.f;
      float w3 = (fabsf(wf.w) * cf.w > th) ? wf.w : 0.f;
      bt[q * 4 + 0] = (__bf16)w0; bt[q * 4 + 1] = (__bf16)w1;
      bt[q * 4 + 2] = (__bf16)w2; bt[q * 4 + 3] = (__bf16)w3;
    }
    v16bf a0 = load_a_bf(xb + (size_t)(0 * 16 + col) * II, kb, half);
    acc0 = wmma_bf16(a0, bt, acc0);
    v16bf a1 = load_a_bf(xb + (size_t)(1 * 16 + col) * II, kb, half);
    acc1 = wmma_bf16(a1, bt, acc1);
    v16bf a2 = load_a_bf(xb + (size_t)(2 * 16 + col) * II, kb, half);
    acc2 = wmma_bf16(a2, bt, acc2);
    v16bf a3 = load_a_bf(xb + (size_t)(3 * 16 + col) * II, kb, half);
    acc3 = wmma_bf16(a3, bt, acc3);
  }

  // Epilogue: C/D layout -> VGPR r holds m = half*8 + r, column = col
  const float ma = mean_astro[o];
  const float bo = bias[o];
  #pragma unroll
  for (int r = 0; r < 8; ++r) {
    int b0 = 0 * 16 + half * 8 + r;
    int b1 = 1 * 16 + half * 8 + r;
    int b2 = 2 * 16 + half * 8 + r;
    int b3 = 3 * 16 + half * 8 + r;
    syn[(size_t)b0 * OO + o] = acc0[r] * ma + bo;
    syn[(size_t)b1 * OO + o] = acc1[r] * ma + bo;
    syn[(size_t)b2 * OO + o] = acc2[r] * ma + bo;
    syn[(size_t)b3 * OO + o] = acc3[r] * ma + bo;
  }
}

// ---------------------------------------------------------------------------
// Kernel 4: dendritic GEMM + fused s-reduction + final ReLU.
//   n = o*10 + s flattened; workgroup = 8 neurons = 80 cols = 5 N-tiles,
//   5 waves; wave -> one N-tile, all 4 M-tiles (168MB D stream read once).
// ---------------------------------------------------------------------------
__global__ void dend_gemm_kernel(const __bf16* __restrict__ xb,
                                 const float* __restrict__ dseg,
                                 const float* __restrict__ gates,
                                 const float* __restrict__ syn,
                                 float* __restrict__ out) {
  __shared__ float lds[BB * 80];  // 20 KB

  const int lane = threadIdx.x & 31;
  const int wv   = threadIdx.x >> 5;  // 0..4
  const int col  = lane & 15;
  const int half = lane >> 4;

  const int n = blockIdx.x * 80 + wv * 16 + col;  // global flattened (o,s)
  const int o = n / SS;
  const int s = n % SS;

  const float* bcol = dseg + (size_t)o * (II * SS) + s;  // element k: bcol[k*SS]
  const float sg = 1.0f / (1.0f + expf(-gates[n]));

  v8f acc0 = {}, acc1 = {}, acc2 = {}, acc3 = {};

  for (int kb = 0; kb < II; kb += 32) {
    v16bf bt;
    const int kk = kb + half * 16;
    #pragma unroll
    for (int j = 0; j < 16; ++j)
      bt[j] = (__bf16)__builtin_nontemporal_load(bcol + (size_t)(kk + j) * SS);

    v16bf a0 = load_a_bf(xb + (size_t)(0 * 16 + col) * II, kb, half);
    acc0 = wmma_bf16(a0, bt, acc0);
    v16bf a1 = load_a_bf(xb + (size_t)(1 * 16 + col) * II, kb, half);
    acc1 = wmma_bf16(a1, bt, acc1);
    v16bf a2 = load_a_bf(xb + (size_t)(2 * 16 + col) * II, kb, half);
    acc2 = wmma_bf16(a2, bt, acc2);
    v16bf a3 = load_a_bf(xb + (size_t)(3 * 16 + col) * II, kb, half);
    acc3 = wmma_bf16(a3, bt, acc3);
  }

  // gated = relu(seg)*sig(gate) -> LDS[b][local_col]
  const int lc = wv * 16 + col;  // 0..79
  #pragma unroll
  for (int r = 0; r < 8; ++r) {
    int m = half * 8 + r;
    lds[(0 * 16 + m) * 80 + lc] = fmaxf(acc0[r], 0.f) * sg;
    lds[(1 * 16 + m) * 80 + lc] = fmaxf(acc1[r], 0.f) * sg;
    lds[(2 * 16 + m) * 80 + lc] = fmaxf(acc2[r], 0.f) * sg;
    lds[(3 * 16 + m) * 80 + lc] = fmaxf(acc3[r], 0.f) * sg;
  }
  __syncthreads();

  // Reduce 10 segments per neuron, add synaptic, final ReLU.
  for (int idx = threadIdx.x; idx < BB * 8; idx += blockDim.x) {
    int b  = idx >> 3;
    int oo = idx & 7;
    float dsum = 0.f;
    #pragma unroll
    for (int t = 0; t < SS; ++t) dsum += lds[b * 80 + oo * SS + t];
    int og = blockIdx.x * 8 + oo;
    out[(size_t)b * OO + og] = fmaxf(syn[(size_t)b * OO + og] + dsum, 0.f);
  }
}

// ---------------------------------------------------------------------------
extern "C" void kernel_launch(void* const* d_in, const int* in_sizes, int n_in,
                              void* d_out, int out_size, void* d_ws, size_t ws_size,
                              hipStream_t stream) {
  const float* x        = (const float*)d_in[0];
  const float* context  = (const float*)d_in[1];
  // d_in[2] = prev_activation (unused by reference)
  const float* weight   = (const float*)d_in[3];
  const float* bias     = (const float*)d_in[4];
  const float* a_act    = (const float*)d_in[5];
  const float* a_thr    = (const float*)d_in[6];
  const float* dseg     = (const float*)d_in[7];
  const float* gates    = (const float*)d_in[8];
  const float* cstr     = (const float*)d_in[9];
  const float* p_thr    = (const float*)d_in[10];

  float* out = (float*)d_out;

  // Workspace layout: syn[B*O] f32, ctx_mean[B] f32, mean_astro[O] f32,
  //                   xb[B*I] bf16 (16B-aligned).
  float*  ws_syn   = (float*)d_ws;
  float*  ws_ctx   = ws_syn + (size_t)BB * OO;
  float*  ws_astro = ws_ctx + BB;
  __bf16* ws_xb    = (__bf16*)(ws_astro + OO);

  pack_x_kernel<<<dim3((BB * II) / (256 * 8)), dim3(256), 0, stream>>>(x, ws_xb);
  ctx_mean_kernel<<<dim3(1), dim3(64), 0, stream>>>(context, ws_ctx);
  astro_kernel<<<dim3(OO / 256), dim3(256), 0, stream>>>(ws_ctx, a_act, a_thr, ws_astro);
  syn_gemm_kernel<<<dim3(OO / 64), dim3(128), 0, stream>>>(
      ws_xb, weight, cstr, p_thr, bias, ws_astro, ws_syn);
  dend_gemm_kernel<<<dim3(OO / 8), dim3(160), 0, stream>>>(
      ws_xb, dseg, gates, ws_syn, out);
}
